// Cffn_88785563943695
// MI455X (gfx1250) — matrix operand order repaired and checked by reading, not compiled, not run on hardware
//
#include <hip/hip_runtime.h>
#include <hip/hip_bf16.h>
#include <cstdint>

// ---------------------------------------------------------------------------
//   linear_out = x @ U_w^T ; gated = sigmoid(x @ G_w^T)*x
//   out = linear_out + sum_l cfrac(gated * ladder_w[l]) * V[:,l]
//
// Pipeline:
//   (1) prep kernel: split fp32 x/U/G into hi/lo bf16 arrays in d_ws (once).
//   (2) GEMM kernel: double-buffered LDS fed by GLOBAL_LOAD_ASYNC_TO_LDS_B128
//       (ASYNCcnt), 3x v_wmma_f32_16x16x32_bf16 per tile (hh+hl+lh split
//       products) accumulating fp32 -> near-fp32 accuracy at bf16 rate.
// Fallback fused kernel (round-1 style) if ws_size is too small.
// ---------------------------------------------------------------------------

typedef __attribute__((ext_vector_type(16))) __bf16 bf16x16;
typedef __attribute__((ext_vector_type(8)))  float  f32x8;
typedef __attribute__((ext_vector_type(4)))  int    i32x4;

#define DIM        1024
#define M_TOTAL    4096      // 2 * 2048
#define BM         128
#define BN         64
#define BK         32
#define KSTEPS     (DIM / BK)   // 32
#define NUM_LADDERS 3
#define DEPTH      5
#define EPSILON    0.01f

#define GLOBAL_AS __attribute__((address_space(1)))
#define LDS_AS    __attribute__((address_space(3)))

union FragAB { uint4 q[2]; bf16x16 v; };
union FragC  { f32x8 v; float f[8]; };

static __device__ __forceinline__ unsigned short f2bf(float f) {
    unsigned int u = __float_as_uint(f);
    unsigned int r = (u + 0x7FFFu + ((u >> 16) & 1u)) >> 16;  // RNE
    return (unsigned short)r;
}
static __device__ __forceinline__ float bf2f(unsigned short h) {
    return __uint_as_float(((unsigned int)h) << 16);
}
static __device__ __forceinline__ void split_store(float v, unsigned short* ph,
                                                   unsigned short* pl) {
    unsigned short h = f2bf(v);
    *ph = h;
    *pl = f2bf(v - bf2f(h));
}

// async 16-byte global -> LDS copy (tracked by ASYNCcnt, no VGPR data)
static __device__ __forceinline__ void async_copy16(const unsigned short* g,
                                                    unsigned short* l) {
    __builtin_amdgcn_global_load_async_to_lds_b128(
        (GLOBAL_AS i32x4*)(uintptr_t)g,
        (LDS_AS i32x4*)(unsigned int)(uintptr_t)l, 0, 0);
}

// guarded continued fraction, depth 5, a_k = gx * w[k]
static __device__ __forceinline__ float cfrac(float gx, const float* w) {
    float z = gx * w[DEPTH - 1];
#pragma unroll
    for (int k = DEPTH - 2; k >= 0; --k) {
        float den  = 1.0f + z;
        float ad   = fabsf(den);
        float safe = (ad < EPSILON) ? ((den >= 0.0f) ? EPSILON : -EPSILON) : den;
        z = (gx * w[k]) / safe;
    }
    return z;
}

// ---------------------------------------------------------------------------
// (1) fp32 -> (hi,lo) bf16 split, 4 elements / thread
// ---------------------------------------------------------------------------
__global__ __launch_bounds__(256)
void split_kernel(const float* __restrict__ src,
                  unsigned short* __restrict__ dh,
                  unsigned short* __restrict__ dl, int n4) {
    int i = blockIdx.x * 256 + threadIdx.x;
    if (i >= n4) return;
    float4 v = ((const float4*)src)[i];
    ushort4 h, l;
    h.x = f2bf(v.x); l.x = f2bf(v.x - bf2f(h.x));
    h.y = f2bf(v.y); l.y = f2bf(v.y - bf2f(h.y));
    h.z = f2bf(v.z); l.z = f2bf(v.z - bf2f(h.z));
    h.w = f2bf(v.w); l.w = f2bf(v.w - bf2f(h.w));
    *(ushort4*)(dh + 4 * (size_t)i) = h;
    *(ushort4*)(dl + 4 * (size_t)i) = l;
}

// ---------------------------------------------------------------------------
// (2) main GEMM + epilogue, async-LDS double buffered
// ---------------------------------------------------------------------------
__global__ __launch_bounds__(256)
void cffn_wmma_async(const unsigned short* __restrict__ xh,
                     const unsigned short* __restrict__ xl,
                     const unsigned short* __restrict__ uh,
                     const unsigned short* __restrict__ ul,
                     const unsigned short* __restrict__ gh,
                     const unsigned short* __restrict__ gl,
                     const float* __restrict__ x,
                     const float* __restrict__ ladder,   // (3, 1024, 5)
                     const float* __restrict__ V,        // (1024, 3)
                     float* __restrict__ out) {
    __shared__ __align__(16) unsigned short sXh[2][BM * BK];
    __shared__ __align__(16) unsigned short sXl[2][BM * BK];
    __shared__ __align__(16) unsigned short sUh[2][BN * BK];
    __shared__ __align__(16) unsigned short sUl[2][BN * BK];
    __shared__ __align__(16) unsigned short sGh[2][BN * BK];
    __shared__ __align__(16) unsigned short sGl[2][BN * BK];

    const int t      = threadIdx.x;
    const int lane   = t & 31;
    const int hlf    = lane >> 4;
    const int lm     = lane & 15;
    const int waveId = t >> 5;
    const int waveM  = waveId & 3;
    const int waveN  = waveId >> 2;

    const int m0 = blockIdx.x * BM;
    const int n0 = blockIdx.y * BN;

    // issue one stage of async copies (8 x 16B per thread)
    auto stage = [&](int kt, int b) {
        const int k0 = kt * BK;
        // x tiles: 128x32 bf16 = 512 chunks of 16B, two arrays
#pragma unroll
        for (int j = 0; j < 2; ++j) {
            int c   = t + j * 256;        // 0..511
            int r   = c >> 2;             // row (4 chunks per 64B row)
            int col = (c & 3) << 3;       // element offset (8 ushorts = 16B)
            size_t go = (size_t)(m0 + r) * DIM + k0 + col;
            int    lo = r * BK + col;
            async_copy16(xh + go, &sXh[b][lo]);
            async_copy16(xl + go, &sXl[b][lo]);
        }
        // weight tiles: 64x32 bf16 = 256 chunks, four arrays
        {
            int r   = t >> 2;
            int col = (t & 3) << 3;
            size_t go = (size_t)(n0 + r) * DIM + k0 + col;
            int    lo = r * BK + col;
            async_copy16(uh + go, &sUh[b][lo]);
            async_copy16(ul + go, &sUl[b][lo]);
            async_copy16(gh + go, &sGh[b][lo]);
            async_copy16(gl + go, &sGl[b][lo]);
        }
    };

    f32x8 accU[2][2], accG[2][2];
#pragma unroll
    for (int a = 0; a < 2; ++a)
#pragma unroll
        for (int b = 0; b < 2; ++b) { accU[a][b] = (f32x8)0.0f; accG[a][b] = (f32x8)0.0f; }

    stage(0, 0);

    for (int kt = 0; kt < KSTEPS; ++kt) {
        const int b = kt & 1;
        if (kt + 1 < KSTEPS) {
            stage(kt + 1, b ^ 1);
            // this wave's previous-stage 8 copies are done once <=8 remain
            asm volatile("s_wait_asynccnt 8" ::: "memory");
        } else {
            asm volatile("s_wait_asynccnt 0" ::: "memory");
        }
        __syncthreads();   // all waves' stage-kt data visible in LDS

        // A frags (16x32 bf16): lanes 0-15 M=lane, K {0..7,16..23};
        //                       lanes16-31 M=lane-16, K {8..15,24..31}
        FragAB ah[2], al[2];
#pragma unroll
        for (int tm = 0; tm < 2; ++tm) {
            const unsigned short* rh = &sXh[b][(waveM * 32 + tm * 16 + lm) * BK];
            const unsigned short* rl = &sXl[b][(waveM * 32 + tm * 16 + lm) * BK];
            int kbase = hlf * 8;
            ah[tm].q[0] = *(const uint4*)(rh + kbase);
            ah[tm].q[1] = *(const uint4*)(rh + kbase + 16);
            al[tm].q[0] = *(const uint4*)(rl + kbase);
            al[tm].q[1] = *(const uint4*)(rl + kbase + 16);
        }
        // B frags (32x16 bf16): lanes 0-15 K {0..7,16..23}, lanes 16-31 K {8..15,24..31}, N=lane&15
#pragma unroll
        for (int tn = 0; tn < 2; ++tn) {
            const int nrow = (waveN * 32 + tn * 16 + lm) * BK + hlf * 8;
            FragAB bh, bl;
            bh.q[0] = *(const uint4*)(&sUh[b][nrow]);
            bh.q[1] = *(const uint4*)(&sUh[b][nrow + 16]);
            bl.q[0] = *(const uint4*)(&sUl[b][nrow]);
            bl.q[1] = *(const uint4*)(&sUl[b][nrow + 16]);
#pragma unroll
            for (int tm = 0; tm < 2; ++tm) {
                accU[tm][tn] = __builtin_amdgcn_wmma_f32_16x16x32_bf16(
                    false, ah[tm].v, false, bh.v, (short)0, accU[tm][tn], false, false);
                accU[tm][tn] = __builtin_amdgcn_wmma_f32_16x16x32_bf16(
                    false, ah[tm].v, false, bl.v, (short)0, accU[tm][tn], false, false);
                accU[tm][tn] = __builtin_amdgcn_wmma_f32_16x16x32_bf16(
                    false, al[tm].v, false, bh.v, (short)0, accU[tm][tn], false, false);
            }
            bh.q[0] = *(const uint4*)(&sGh[b][nrow]);
            bh.q[1] = *(const uint4*)(&sGh[b][nrow + 16]);
            bl.q[0] = *(const uint4*)(&sGl[b][nrow]);
            bl.q[1] = *(const uint4*)(&sGl[b][nrow + 16]);
#pragma unroll
            for (int tm = 0; tm < 2; ++tm) {
                accG[tm][tn] = __builtin_amdgcn_wmma_f32_16x16x32_bf16(
                    false, ah[tm].v, false, bh.v, (short)0, accG[tm][tn], false, false);
                accG[tm][tn] = __builtin_amdgcn_wmma_f32_16x16x32_bf16(
                    false, ah[tm].v, false, bl.v, (short)0, accG[tm][tn], false, false);
                accG[tm][tn] = __builtin_amdgcn_wmma_f32_16x16x32_bf16(
                    false, al[tm].v, false, bh.v, (short)0, accG[tm][tn], false, false);
            }
        }
        __syncthreads();   // safe to overwrite buffer b next iteration
    }

    // epilogue: C layout -> VGPR i, lanes 0-15 (M=i,N=lane); lanes 16-31 (M=8+i,N=lane-16)
    const int m_base = m0 + waveM * 32;
    const int n_base = n0 + waveN * 32;
#pragma unroll
    for (int tn = 0; tn < 2; ++tn) {
        const int n = n_base + tn * 16 + lm;
        float wv[NUM_LADDERS][DEPTH];
        float vv[NUM_LADDERS];
#pragma unroll
        for (int l = 0; l < NUM_LADDERS; ++l) {
#pragma unroll
            for (int k = 0; k < DEPTH; ++k)
                wv[l][k] = ladder[((size_t)l * DIM + n) * DEPTH + k];
            vv[l] = V[(size_t)n * NUM_LADDERS + l];
        }
#pragma unroll
        for (int tm = 0; tm < 2; ++tm) {
            FragC cu, cg;
            cu.v = accU[tm][tn];
            cg.v = accG[tm][tn];
#pragma unroll
            for (int i = 0; i < 8; ++i) {
                const int m = m_base + tm * 16 + hlf * 8 + i;
                const float xe = x[(size_t)m * DIM + n];
                const float s  = 1.0f / (1.0f + __expf(-cg.f[i]));
                const float gx = s * xe;
                float comb = 0.0f;
#pragma unroll
                for (int l = 0; l < NUM_LADDERS; ++l)
                    comb = fmaf(cfrac(gx, wv[l]), vv[l], comb);
                out[(size_t)m * DIM + n] = cu.f[i] + comb;
            }
        }
    }
}

// ---------------------------------------------------------------------------
// fallback: round-1 fully fused kernel (on-the-fly split), used if ws too small
// ---------------------------------------------------------------------------
__global__ __launch_bounds__(256)
void cffn_fused_kernel(const float* __restrict__ x,
                       const float* __restrict__ Uw,
                       const float* __restrict__ Gw,
                       const float* __restrict__ ladder,
                       const float* __restrict__ V,
                       float* __restrict__ out) {
    __shared__ __align__(16) unsigned short sXh[BM * BK];
    __shared__ __align__(16) unsigned short sXl[BM * BK];
    __shared__ __align__(16) unsigned short sUh[BN * BK];
    __shared__ __align__(16) unsigned short sUl[BN * BK];
    __shared__ __align__(16) unsigned short sGh[BN * BK];
    __shared__ __align__(16) unsigned short sGl[BN * BK];

    const int t      = threadIdx.x;
    const int lane   = t & 31;
    const int hlf    = lane >> 4;
    const int lm     = lane & 15;
    const int waveId = t >> 5;
    const int waveM  = waveId & 3;
    const int waveN  = waveId >> 2;
    const int m0 = blockIdx.x * BM;
    const int n0 = blockIdx.y * BN;

    f32x8 accU[2][2], accG[2][2];
#pragma unroll
    for (int a = 0; a < 2; ++a)
#pragma unroll
        for (int b = 0; b < 2; ++b) { accU[a][b] = (f32x8)0.0f; accG[a][b] = (f32x8)0.0f; }

    for (int kt = 0; kt < KSTEPS; ++kt) {
        const int k0 = kt * BK;
        __syncthreads();
#pragma unroll
        for (int j = 0; j < 4; ++j) {
            int idx = t + j * 256;
            int r = idx >> 3;
            int c = (idx & 7) << 2;
            float4 v = *(const float4*)(x + (size_t)(m0 + r) * DIM + k0 + c);
            split_store(v.x, &sXh[r * BK + c + 0], &sXl[r * BK + c + 0]);
            split_store(v.y, &sXh[r * BK + c + 1], &sXl[r * BK + c + 1]);
            split_store(v.z, &sXh[r * BK + c + 2], &sXl[r * BK + c + 2]);
            split_store(v.w, &sXh[r * BK + c + 3], &sXl[r * BK + c + 3]);
        }
#pragma unroll
        for (int j = 0; j < 2; ++j) {
            int idx = t + j * 256;
            int r = idx >> 3;
            int c = (idx & 7) << 2;
            size_t off = (size_t)(n0 + r) * DIM + k0 + c;
            float4 uv = *(const float4*)(Uw + off);
            split_store(uv.x, &sUh[r * BK + c + 0], &sUl[r * BK + c + 0]);
            split_store(uv.y, &sUh[r * BK + c + 1], &sUl[r * BK + c + 1]);
            split_store(uv.z, &sUh[r * BK + c + 2], &sUl[r * BK + c + 2]);
            split_store(uv.w, &sUh[r * BK + c + 3], &sUl[r * BK + c + 3]);
            float4 gv = *(const float4*)(Gw + off);
            split_store(gv.x, &sGh[r * BK + c + 0], &sGl[r * BK + c + 0]);
            split_store(gv.y, &sGh[r * BK + c + 1], &sGl[r * BK + c + 1]);
            split_store(gv.z, &sGh[r * BK + c + 2], &sGl[r * BK + c + 2]);
            split_store(gv.w, &sGh[r * BK + c + 3], &sGl[r * BK + c + 3]);
        }
        __syncthreads();

        FragAB ah[2], al[2];
#pragma unroll
        for (int tm = 0; tm < 2; ++tm) {
            const unsigned short* rh = &sXh[(waveM * 32 + tm * 16 + lm) * BK];
            const unsigned short* rl = &sXl[(waveM * 32 + tm * 16 + lm) * BK];
            int kbase = hlf * 8;
            ah[tm].q[0] = *(const uint4*)(rh + kbase);
            ah[tm].q[1] = *(const uint4*)(rh + kbase + 16);
            al[tm].q[0] = *(const uint4*)(rl + kbase);
            al[tm].q[1] = *(const uint4*)(rl + kbase + 16);
        }
#pragma unroll
        for (int tn = 0; tn < 2; ++tn) {
            const int nrow = (waveN * 32 + tn * 16 + lm) * BK + hlf * 8;
            FragAB bh, bl;
            bh.q[0] = *(const uint4*)(&sUh[nrow]);
            bh.q[1] = *(const uint4*)(&sUh[nrow + 16]);
            bl.q[0] = *(const uint4*)(&sUl[nrow]);
            bl.q[1] = *(const uint4*)(&sUl[nrow + 16]);
#pragma unroll
            for (int tm = 0; tm < 2; ++tm) {
                accU[tm][tn] = __builtin_amdgcn_wmma_f32_16x16x32_bf16(
                    false, ah[tm].v, false, bh.v, (short)0, accU[tm][tn], false, false);
                accU[tm][tn] = __builtin_amdgcn_wmma_f32_16x16x32_bf16(
                    false, ah[tm].v, false, bl.v, (short)0, accU[tm][tn], false, false);
                accU[tm][tn] = __builtin_amdgcn_wmma_f32_16x16x32_bf16(
                    false, al[tm].v, false, bh.v, (short)0, accU[tm][tn], false, false);
            }
            bh.q[0] = *(const uint4*)(&sGh[nrow]);
            bh.q[1] = *(const uint4*)(&sGh[nrow + 16]);
            bl.q[0] = *(const uint4*)(&sGl[nrow]);
            bl.q[1] = *(const uint4*)(&sGl[nrow + 16]);
#pragma unroll
            for (int tm = 0; tm < 2; ++tm) {
                accG[tm][tn] = __builtin_amdgcn_wmma_f32_16x16x32_bf16(
                    false, ah[tm].v, false, bh.v, (short)0, accG[tm][tn], false, false);
                accG[tm][tn] = __builtin_amdgcn_wmma_f32_16x16x32_bf16(
                    false, ah[tm].v, false, bl.v, (short)0, accG[tm][tn], false, false);
                accG[tm][tn] = __builtin_amdgcn_wmma_f32_16x16x32_bf16(
                    false, al[tm].v, false, bh.v, (short)0, accG[tm][tn], false, false);
            }
        }
    }

    const int m_base = m0 + waveM * 32;
    const int n_base = n0 + waveN * 32;
#pragma unroll
    for (int tn = 0; tn < 2; ++tn) {
        const int n = n_base + tn * 16 + lm;
        float wv[NUM_LADDERS][DEPTH];
        float vv[NUM_LADDERS];
#pragma unroll
        for (int l = 0; l < NUM_LADDERS; ++l) {
#pragma unroll
            for (int k = 0; k < DEPTH; ++k)
                wv[l][k] = ladder[((size_t)l * DIM + n) * DEPTH + k];
            vv[l] = V[(size_t)n * NUM_LADDERS + l];
        }
#pragma unroll
        for (int tm = 0; tm < 2; ++tm) {
            FragC cu, cg;
            cu.v = accU[tm][tn];
            cg.v = accG[tm][tn];
#pragma unroll
            for (int i = 0; i < 8; ++i) {
                const int m = m_base + tm * 16 + hlf * 8 + i;
                const float xe = x[(size_t)m * DIM + n];
                const float s  = 1.0f / (1.0f + __expf(-cg.f[i]));
                const float gx = s * xe;
                float comb = 0.0f;
#pragma unroll
                for (int l = 0; l < NUM_LADDERS; ++l)
                    comb = fmaf(cfrac(gx, wv[l]), vv[l], comb);
                out[(size_t)m * DIM + n] = cu.f[i] + comb;
            }
        }
    }
}

extern "C" void kernel_launch(void* const* d_in, const int* in_sizes, int n_in,
                              void* d_out, int out_size, void* d_ws, size_t ws_size,
                              hipStream_t stream) {
    const float* x      = (const float*)d_in[0];   // (2, 2048, 1024)
    const float* U_w    = (const float*)d_in[1];   // (1024, 1024)
    const float* gate_w = (const float*)d_in[2];   // (1024, 1024)
    const float* ladder = (const float*)d_in[3];   // (3, 1024, 5)
    const float* V      = (const float*)d_in[4];   // (1024, 3)
    float* out          = (float*)d_out;

    const size_t NX = (size_t)M_TOTAL * DIM;       // 4 Mi elements
    const size_t NW = (size_t)DIM * DIM;           // 1 Mi elements
    const size_t need = (2 * NX + 4 * NW) * sizeof(unsigned short);  // 24 MB

    dim3 grid(M_TOTAL / BM, DIM / BN);             // (32, 16)
    dim3 block(256);

    if (ws_size >= need) {
        unsigned short* base = (unsigned short*)d_ws;
        unsigned short* xh = base;
        unsigned short* xl = xh + NX;
        unsigned short* uh = xl + NX;
        unsigned short* ul = uh + NW;
        unsigned short* gh = ul + NW;
        unsigned short* gl = gh + NW;

        split_kernel<<<(unsigned)(NX / 4 + 255) / 256, 256, 0, stream>>>(x, xh, xl, (int)(NX / 4));
        split_kernel<<<(unsigned)(NW / 4 + 255) / 256, 256, 0, stream>>>(U_w, uh, ul, (int)(NW / 4));
        split_kernel<<<(unsigned)(NW / 4 + 255) / 256, 256, 0, stream>>>(gate_w, gh, gl, (int)(NW / 4));

        cffn_wmma_async<<<grid, block, 0, stream>>>(xh, xl, uh, ul, gh, gl,
                                                    x, ladder, V, out);
    } else {
        cffn_fused_kernel<<<grid, block, 0, stream>>>(x, U_w, gate_w, ladder, V, out);
    }
}
